// MolOptDecoder_25555055411370
// MI455X (gfx1250) — matrix operand order repaired and checked by reading, not compiled, not run on hardware
//
#include <hip/hip_runtime.h>
#include <hip/hip_bf16.h>

// ---------------------------------------------------------------------------
// MolOptDecoder on MI455X (gfx1250): bf16 WMMA with f32 accumulation.
// Working set ~37 MB -> fully L2 resident (192 MB). HBM floor ~1us.
// ---------------------------------------------------------------------------

typedef __attribute__((ext_vector_type(16))) __bf16 v16bf;
typedef __attribute__((ext_vector_type(8)))  __bf16 v8bf;
typedef __attribute__((ext_vector_type(8)))  float  v8f;

#define NEG_SLOPE 0.01f

// Problem constants
#define ROWS   10240      // B*N
#define DD     256        // D == H
#define NS     40
#define NC     5
#define NB     5
#define NMOL   256
#define NATOM  40

// Workspace layout (bytes)
#define OFF_XB   ((size_t)0)                 // 10240*256 bf16  = 5,242,880
#define OFF_HS   ((size_t)5242880)           // 10240*256 bf16
#define OFF_HC   ((size_t)10485760)          // 10240*256 bf16
#define OFF_P1   ((size_t)15728640)          // 10240*256 f32   = 10,485,760
#define OFF_P2   ((size_t)26214400)          // 10240*256 f32
#define OFF_W1B  ((size_t)36700160)          // 8*64*32*16 bf16 = 524,288
#define OFF_W2B  ((size_t)37224448)          // 8*4*32*16 bf16  = 32,768
#define OFF_WBB  ((size_t)37257216)          // 8*32*16 bf16    = 8,192
// total: 37,265,408 bytes

static __device__ __forceinline__ float lrelu(float v) {
    return v >= 0.0f ? v : NEG_SLOPE * v;
}

// ---------------------------------------------------------------------------
// Kernel 1: convert yhat -> bf16, and pre-swizzle weights into the exact
// per-lane WMMA B-fragment layout (32x16 bf16 tile):
//   fragment (kstep s, ntile t), lane l, element e  <=
//     W[k = s*32 + ((l&16)?8:0) + e + ((e&8)?8:0)][n = t*16 + (l&15)]
// W1 = concat_n( Ws1 | Wc1 | Wb1[:D] | Wb1[D:] )  -> [256, 1024]
// W2 = pad64( Ws2 (n 0..39) | zeros | Wc2 (n 48..52) | zeros )
// Wb = pad16( Wb2 (n 0..4) | zeros )
// ---------------------------------------------------------------------------
__global__ void molopt_prep(const float* __restrict__ yhat,
                            const float* __restrict__ Ws1,
                            const float* __restrict__ Wc1,
                            const float* __restrict__ Wb1,
                            const float* __restrict__ Ws2,
                            const float* __restrict__ Wc2,
                            const float* __restrict__ Wb2,
                            __bf16* __restrict__ xb,
                            __bf16* __restrict__ W1B,
                            __bf16* __restrict__ W2B,
                            __bf16* __restrict__ WbB)
{
    const int XB_N = ROWS * DD;        // 2,621,440
    const int W1_N = 8 * 64 * 32 * 16; // 262,144
    const int W2_N = 8 * 4 * 32 * 16;  // 16,384
    const int WB_N = 8 * 32 * 16;      // 4,096
    const int TOT  = XB_N + W1_N + W2_N + WB_N;

    for (int idx = blockIdx.x * blockDim.x + threadIdx.x; idx < TOT;
         idx += gridDim.x * blockDim.x) {
        if (idx < XB_N) { xb[idx] = (__bf16)yhat[idx]; continue; }
        int id = idx - XB_N;
        if (id < W1_N) {
            int e = id & 15, l = (id >> 4) & 31, t = (id >> 9) & 63, s = id >> 15;
            int k = s * 32 + ((l & 16) ? 8 : 0) + e + ((e & 8) ? 8 : 0);
            int n = t * 16 + (l & 15);
            float v;
            if (n < 256)      v = Ws1[k * 256 + n];
            else if (n < 512) v = Wc1[k * 256 + (n - 256)];
            else if (n < 768) v = Wb1[k * 256 + (n - 512)];
            else              v = Wb1[(256 + k) * 256 + (n - 768)];
            W1B[id] = (__bf16)v;
            continue;
        }
        id -= W1_N;
        if (id < W2_N) {
            int e = id & 15, l = (id >> 4) & 31, t = (id >> 9) & 3, s = id >> 11;
            int k = s * 32 + ((l & 16) ? 8 : 0) + e + ((e & 8) ? 8 : 0);
            int n = t * 16 + (l & 15);
            float v = 0.0f;
            if (n < 40)                 v = Ws2[k * NS + n];
            else if (n >= 48 && n < 53) v = Wc2[k * NC + (n - 48)];
            W2B[id] = (__bf16)v;
            continue;
        }
        id -= W2_N;
        {
            int e = id & 15, l = (id >> 4) & 31, s = id >> 9;
            int k = s * 32 + ((l & 16) ? 8 : 0) + e + ((e & 8) ? 8 : 0);
            int n = l & 15;
            WbB[id] = (__bf16)((n < NB) ? Wb2[k * NB + n] : 0.0f);
        }
    }
}

// ---------------------------------------------------------------------------
// Kernel 2: fused GEMM  [10240,256] x [256,1024]
//   n-section 0: HS = lrelu(. + bs1) (bf16)   1: HC = lrelu(. + bc1) (bf16)
//   n-section 2: P1 = .               (f32)   3: P2 = . + bb1        (f32)
// One wave per 16x16 output tile; 8 WMMA k-steps (K=256).
// A fragment (16-bit A 16x32): lane<16 -> K {ks..ks+7, ks+16..ks+23},
// lane>=16 -> +8. Contiguous 16B loads from row-major bf16 X.
// ---------------------------------------------------------------------------
__global__ __launch_bounds__(256) void molopt_gemm1(
    const __bf16* __restrict__ xb, const __bf16* __restrict__ W1B,
    const float* __restrict__ bs1, const float* __restrict__ bc1,
    const float* __restrict__ bb1,
    __bf16* __restrict__ HS, __bf16* __restrict__ HC,
    float* __restrict__ P1, float* __restrict__ P2)
{
    const int lane = threadIdx.x & 31;
    const int wid  = (blockIdx.x * blockDim.x + threadIdx.x) >> 5;
    const int mt   = wid >> 6;   // 0..639
    const int nt   = wid & 63;   // 0..63

    const int row = mt * 16 + (lane & 15);
    const __bf16* abase = xb + (size_t)row * DD + ((lane & 16) ? 8 : 0);

    v8f acc = {0.f, 0.f, 0.f, 0.f, 0.f, 0.f, 0.f, 0.f};
#pragma unroll
    for (int s = 0; s < 8; ++s) {
        const v8bf lo = *(const v8bf*)(abase + s * 32);
        const v8bf hi = *(const v8bf*)(abase + s * 32 + 16);
        v16bf A;
#pragma unroll
        for (int e = 0; e < 8; ++e) { A[e] = lo[e]; A[8 + e] = hi[e]; }
        const v16bf Bf = *(const v16bf*)(W1B + ((((s * 64 + nt) * 32) + lane) << 4));
        acc = __builtin_amdgcn_wmma_f32_16x16x32_bf16(false, A, false, Bf,
                                                      (short)0, acc, false, false);
    }

    const int sec = nt >> 4;
    const int col = (nt & 15) * 16 + (lane & 15);
    const int m0  = (lane & 16) ? 8 : 0;
#pragma unroll
    for (int r = 0; r < 8; ++r) {
        const int gr = mt * 16 + m0 + r;
        const size_t o = (size_t)gr * DD + col;
        float v = acc[r];
        if (sec == 0)      { v = lrelu(v + bs1[col]); HS[o] = (__bf16)v; }
        else if (sec == 1) { v = lrelu(v + bc1[col]); HC[o] = (__bf16)v; }
        else if (sec == 2) { P1[o] = v; }
        else               { P2[o] = v + bb1[col]; }   // fold bb1 into P2
    }
}

// ---------------------------------------------------------------------------
// Kernel 3: heads  symbols = HS @ Ws2 + bs2 (N=40, tiles 0..2)
//                  charges = HC @ Wc2 + bc2 (N=5, padded to tile 3 cols 48..52)
// ---------------------------------------------------------------------------
__global__ __launch_bounds__(256) void molopt_gemm2(
    const __bf16* __restrict__ HS, const __bf16* __restrict__ HC,
    const __bf16* __restrict__ W2B,
    const float* __restrict__ bs2, const float* __restrict__ bc2,
    float* __restrict__ symbols, float* __restrict__ charges)
{
    const int lane = threadIdx.x & 31;
    const int wid  = (blockIdx.x * blockDim.x + threadIdx.x) >> 5;
    const int mt   = wid >> 2;   // 0..639
    const int nt   = wid & 3;    // 0..3  (3 == charges tile)

    const __bf16* Xsrc = (nt < 3) ? HS : HC;
    const int row = mt * 16 + (lane & 15);
    const __bf16* abase = Xsrc + (size_t)row * DD + ((lane & 16) ? 8 : 0);

    v8f acc = {0.f, 0.f, 0.f, 0.f, 0.f, 0.f, 0.f, 0.f};
#pragma unroll
    for (int s = 0; s < 8; ++s) {
        const v8bf lo = *(const v8bf*)(abase + s * 32);
        const v8bf hi = *(const v8bf*)(abase + s * 32 + 16);
        v16bf A;
#pragma unroll
        for (int e = 0; e < 8; ++e) { A[e] = lo[e]; A[8 + e] = hi[e]; }
        const v16bf Bf = *(const v16bf*)(W2B + ((((s * 4 + nt) * 32) + lane) << 4));
        acc = __builtin_amdgcn_wmma_f32_16x16x32_bf16(false, A, false, Bf,
                                                      (short)0, acc, false, false);
    }

    const int col = nt * 16 + (lane & 15);   // in padded [0,64)
    const int m0  = (lane & 16) ? 8 : 0;
#pragma unroll
    for (int r = 0; r < 8; ++r) {
        const int gr = mt * 16 + m0 + r;
        if (col < NS) {
            symbols[(size_t)gr * NS + col] = acc[r] + bs2[col];
        } else if (col >= 48 && col < 48 + NC) {
            const int c = col - 48;
            charges[(size_t)gr * NC + c] = acc[r] + bc2[c];
        }
    }
}

// ---------------------------------------------------------------------------
// Kernel 4: bond head. One workgroup (8 waves) per molecule.
// For each pair-tile of 16 rows: build A = bf16(lrelu(P1[i]+P2[j])) on the
// fly (bb1 already folded into P2) and contract K=256 against the padded Wb2
// B fragment with 8 WMMAs. Results go to LDS sT[1600][5]; after a barrier the
// symmetrization pass writes sT[i,j] + sT[j,i] + 2*bb2. The h tensor
// ([256,40,40,256] = 420 MB) never exists in memory.
// ---------------------------------------------------------------------------
__global__ __launch_bounds__(256) void molopt_bonds(
    const float* __restrict__ P1, const float* __restrict__ P2,
    const __bf16* __restrict__ WbB, const float* __restrict__ bb2,
    float* __restrict__ outB)
{
    __shared__ float sT[NATOM * NATOM * NB];   // 1600*5 f32 = 32 KB

    const int b    = blockIdx.x;
    const int t    = threadIdx.x;
    const int lane = t & 31;
    const int wave = t >> 5;

    // Preload all 8 k-step B fragments of Wb2 (zero-padded N=16); reused by
    // every pair-tile this wave processes.
    v16bf Bf[8];
#pragma unroll
    for (int s = 0; s < 8; ++s)
        Bf[s] = *(const v16bf*)(WbB + (((s * 32) + lane) << 4));

    const int   ksel = (lane & 16) ? 8 : 0;
    const float* P1b = P1 + (size_t)b * NATOM * DD;
    const float* P2b = P2 + (size_t)b * NATOM * DD;

    for (int mtb = wave; mtb < 100; mtb += 8) {          // 1600/16 pair tiles
        const int p = mtb * 16 + (lane & 15);
        const int i = p / NATOM, j = p - i * NATOM;
        const float* r1 = P1b + i * DD;
        const float* r2 = P2b + j * DD;

        v8f acc = {0.f, 0.f, 0.f, 0.f, 0.f, 0.f, 0.f, 0.f};
#pragma unroll
        for (int s = 0; s < 8; ++s) {
            const int k0 = s * 32 + ksel;
            const float4 a0 = *(const float4*)(r1 + k0);
            const float4 a1 = *(const float4*)(r1 + k0 + 4);
            const float4 a2 = *(const float4*)(r1 + k0 + 16);
            const float4 a3 = *(const float4*)(r1 + k0 + 20);
            const float4 c0 = *(const float4*)(r2 + k0);
            const float4 c1 = *(const float4*)(r2 + k0 + 4);
            const float4 c2 = *(const float4*)(r2 + k0 + 16);
            const float4 c3 = *(const float4*)(r2 + k0 + 20);
            float h[16] = { a0.x + c0.x, a0.y + c0.y, a0.z + c0.z, a0.w + c0.w,
                            a1.x + c1.x, a1.y + c1.y, a1.z + c1.z, a1.w + c1.w,
                            a2.x + c2.x, a2.y + c2.y, a2.z + c2.z, a2.w + c2.w,
                            a3.x + c3.x, a3.y + c3.y, a3.z + c3.z, a3.w + c3.w };
            v16bf A;
#pragma unroll
            for (int e = 0; e < 16; ++e) A[e] = (__bf16)lrelu(h[e]);
            acc = __builtin_amdgcn_wmma_f32_16x16x32_bf16(false, A, false, Bf[s],
                                                          (short)0, acc, false, false);
        }

        const int n  = lane & 15;
        const int m0 = (lane & 16) ? 8 : 0;
        if (n < NB) {
#pragma unroll
            for (int r = 0; r < 8; ++r) {
                const int pp = mtb * 16 + m0 + r;
                sT[pp * NB + n] = acc[r];
            }
        }
    }
    __syncthreads();

    float b2[NB];
#pragma unroll
    for (int c = 0; c < NB; ++c) b2[c] = 2.0f * bb2[c];

    for (int idx = t; idx < NATOM * NATOM * NB; idx += 256) {
        const int c  = idx % NB;
        const int pj = idx / NB;
        const int i  = pj / NATOM, j = pj - i * NATOM;
        outB[(size_t)b * (NATOM * NATOM * NB) + idx] =
            sT[(i * NATOM + j) * NB + c] + sT[(j * NATOM + i) * NB + c] + b2[c];
    }
}

// ---------------------------------------------------------------------------
extern "C" void kernel_launch(void* const* d_in, const int* in_sizes, int n_in,
                              void* d_out, int out_size, void* d_ws, size_t ws_size,
                              hipStream_t stream)
{
    const float* yhat = (const float*)d_in[0];
    const float* Ws1  = (const float*)d_in[1];
    const float* bs1  = (const float*)d_in[2];
    const float* Ws2  = (const float*)d_in[3];
    const float* bs2  = (const float*)d_in[4];
    const float* Wc1  = (const float*)d_in[5];
    const float* bc1  = (const float*)d_in[6];
    const float* Wc2  = (const float*)d_in[7];
    const float* bc2  = (const float*)d_in[8];
    const float* Wb1  = (const float*)d_in[9];
    const float* bb1  = (const float*)d_in[10];
    const float* Wb2  = (const float*)d_in[11];
    const float* bb2  = (const float*)d_in[12];

    char* ws = (char*)d_ws;
    __bf16* xb  = (__bf16*)(ws + OFF_XB);
    __bf16* HS  = (__bf16*)(ws + OFF_HS);
    __bf16* HC  = (__bf16*)(ws + OFF_HC);
    float*  P1  = (float*)(ws + OFF_P1);
    float*  P2  = (float*)(ws + OFF_P2);
    __bf16* W1B = (__bf16*)(ws + OFF_W1B);
    __bf16* W2B = (__bf16*)(ws + OFF_W2B);
    __bf16* WbB = (__bf16*)(ws + OFF_WBB);

    float* symbols = (float*)d_out;            // [10240, 40]
    float* charges = symbols + ROWS * NS;      // [10240, 5]
    float* bondsO  = charges + ROWS * NC;      // [256*1600, 5]

    molopt_prep <<<2048, 256, 0, stream>>>(yhat, Ws1, Wc1, Wb1, Ws2, Wc2, Wb2,
                                           xb, W1B, W2B, WbB);
    molopt_gemm1<<<5120, 256, 0, stream>>>(xb, W1B, bs1, bc1, bb1, HS, HC, P1, P2);
    molopt_gemm2<<< 320, 256, 0, stream>>>(HS, HC, W2B, bs2, bc2, symbols, charges);
    molopt_bonds<<< 256, 256, 0, stream>>>(P1, P2, WbB, bb2, bondsO);
}
